// HetGNN_49752901157170
// MI455X (gfx1250) — matrix operand order repaired and checked by reading, not compiled, not run on hardware
//
#include <hip/hip_runtime.h>
#include <hip/hip_bf16.h>
#include <math.h>

typedef _Float16 f16;
typedef __attribute__((ext_vector_type(16))) _Float16 v16h;
typedef __attribute__((ext_vector_type(8)))  float    v8f;

#define LH    32
#define FEAT  6
#define NTILE 8   /* 128 gates / 16 */

// Branch-free hardware transcendentals: v_exp_f32 (native exp2) + v_rcp_f32.
// These are TRANS32 ops that co-execute with XDL WMMA.
__device__ __forceinline__ float sigmoidf_(float x) {
  return __builtin_amdgcn_rcpf(
      1.0f + __builtin_amdgcn_exp2f(-1.442695040888963f * x));
}
__device__ __forceinline__ float tanhf_(float x) {
  return fmaf(2.0f,
              __builtin_amdgcn_rcpf(
                  1.0f + __builtin_amdgcn_exp2f(-2.885390081777927f * x)),
              -1.0f);
}

__device__ __forceinline__ v8f wmma_f16(v16h a, v16h b, v8f c) {
  return __builtin_amdgcn_wmma_f32_16x16x32_f16(false, a, false, b, (short)0, c,
                                                false, false);
}

// ---------------------------------------------------------------------------
// Fused 2-layer LSTM. One wave == one tile of 16 channels.
// A-fragment:   lane l -> row M=l&15 ; half j -> K=(j<8?j:j+8)+8*(l>>4)
// B-fragment:   lane l -> col N=l&15 ; half j -> K=(j<8?j:j+8)+8*(l>>4)
// C/D fragment: vgpr r -> row M=r+8*(l>>4) ; col N=l&15
// ---------------------------------------------------------------------------
__global__ __launch_bounds__(128) void lstm2_wmma_kernel(
    const float* __restrict__ cts,                      // (NC, T, 6)
    const float* __restrict__ Wih0, const float* __restrict__ Whh0,
    const float* __restrict__ bih0, const float* __restrict__ bhh0,
    const float* __restrict__ Wih1, const float* __restrict__ Whh1,
    const float* __restrict__ bih1, const float* __restrict__ bhh1,
    float* __restrict__ lstm_last,                      // (NC, 32)
    int nc, int tsteps)
{
  __shared__ _Float16 lds_stage[4][16 * LH];            // 1KB per wave
  const int lane = threadIdx.x & 31;
  const int wave = threadIdx.x >> 5;
  const int tile = blockIdx.x * 4 + wave;
  if (tile * 16 >= nc) return;
  const int c0 = tile * 16;
  const int n  = lane & 15;
  const int hi = lane >> 4;
  const int kb = hi * 16;
  _Float16* stage = &lds_stage[wave][0];

  // ---- preload all weights as register-resident B fragments ----
  v16h Bx0[NTILE], Bh0[NTILE], Bi1[NTILE], Bh1[NTILE];
  float bias0[NTILE], bias1[NTILE];
  for (int nt = 0; nt < NTILE; ++nt) {
    const int g = nt * 16 + n;
    v16h b0 = {}, b1 = {}, b2 = {}, b3 = {};
#pragma unroll
    for (int j = 0; j < 16; ++j) {
      const int K = ((j < 8) ? j : j + 8) + 8 * hi;
      b0[j] = (K < FEAT) ? (f16)Wih0[g * FEAT + K] : (f16)0.0f;
      b1[j] = (f16)Whh0[g * LH + K];
      b2[j] = (f16)Wih1[g * LH + K];
      b3[j] = (f16)Whh1[g * LH + K];
    }
    Bx0[nt] = b0; Bh0[nt] = b1; Bi1[nt] = b2; Bh1[nt] = b3;
    bias0[nt] = bih0[g] + bhh0[g];
    bias1[nt] = bih1[g] + bhh1[g];
  }

  v16h hA0 = {};            // layer-0 recurrent h (A layout)
  v16h hA1 = {};            // layer-1 recurrent h (A layout)
  v8f  cs0[2] = {};         // layer-0 cell state (C layout, kt halves)
  v8f  cs1[2] = {};

  const float* xrow = cts + (size_t)(c0 + n) * tsteps * FEAT;

  for (int t = 0; t < tsteps; ++t) {
    // A-fragment for x_t : only K=0..5 valid (lanes 16..31 cover K>=8 -> zero)
    v16h ax = {};
    if (hi == 0) {
#pragma unroll
      for (int j = 0; j < FEAT; ++j) ax[j] = (f16)xrow[t * FEAT + j];
    }

    // ---------------- layer 0: gates = x@Wih0^T + h@Whh0^T + bias ----------
    v8f acc[NTILE];
#pragma unroll
    for (int nt = 0; nt < NTILE; ++nt) {
      v8f c;
#pragma unroll
      for (int r = 0; r < 8; ++r) c[r] = bias0[nt];
      c = wmma_f16(ax,  Bx0[nt], c);
      c = wmma_f16(hA0, Bh0[nt], c);
      acc[nt] = c;
    }
    // elementwise: branch-free TRANS32 ops (co-execute with WMMA)
#pragma unroll
    for (int kt = 0; kt < 2; ++kt) {
#pragma unroll
      for (int r = 0; r < 8; ++r) {
        const float iv = sigmoidf_(acc[0 + kt][r]);
        const float fv = sigmoidf_(acc[2 + kt][r]);
        const float gv = tanhf_(acc[4 + kt][r]);
        const float ov = sigmoidf_(acc[6 + kt][r]);
        const float cc = fv * cs0[kt][r] + iv * gv;
        cs0[kt][r] = cc;
        const float hh = ov * tanhf_(cc);
        stage[(r + 8 * hi) * LH + kt * 16 + n] = (f16)hh;   // C->A transpose via LDS
      }
    }
    v16h a1;  // h1 as A fragment: feeds layer-1 input AND layer-0 recurrence
#pragma unroll
    for (int j = 0; j < 16; ++j) {
      const int K = ((j < 8) ? j : j + 8) + 8 * hi;
      a1[j] = stage[n * LH + K];
    }
    hA0 = a1;

    // ---------------- layer 1 ----------------------------------------------
    v8f acc1[NTILE];
#pragma unroll
    for (int nt = 0; nt < NTILE; ++nt) {
      v8f c;
#pragma unroll
      for (int r = 0; r < 8; ++r) c[r] = bias1[nt];
      c = wmma_f16(a1,  Bi1[nt], c);
      c = wmma_f16(hA1, Bh1[nt], c);
      acc1[nt] = c;
    }
#pragma unroll
    for (int kt = 0; kt < 2; ++kt) {
#pragma unroll
      for (int r = 0; r < 8; ++r) {
        const float iv = sigmoidf_(acc1[0 + kt][r]);
        const float fv = sigmoidf_(acc1[2 + kt][r]);
        const float gv = tanhf_(acc1[4 + kt][r]);
        const float ov = sigmoidf_(acc1[6 + kt][r]);
        const float cc = fv * cs1[kt][r] + iv * gv;
        cs1[kt][r] = cc;
        const float hh = ov * tanhf_(cc);
        stage[(r + 8 * hi) * LH + kt * 16 + n] = (f16)hh;
        if (t == tsteps - 1) {
          lstm_last[(size_t)(c0 + r + 8 * hi) * LH + kt * 16 + n] = hh;
        }
      }
    }
    v16h a2;
#pragma unroll
    for (int j = 0; j < 16; ++j) {
      const int K = ((j < 8) ? j : j + 8) + 8 * hi;
      a2[j] = stage[n * LH + K];
    }
    hA1 = a2;
  }
}

// ---------------------------------------------------------------------------
// xc = concat(x_channel[12], lstm_last[32]) @ Wc^T + bc      (one thread/elem)
// ---------------------------------------------------------------------------
__global__ void proj_channel_kernel(const float* __restrict__ xch,
                                    const float* __restrict__ lstm_last,
                                    const float* __restrict__ W,   // (16,44)
                                    const float* __restrict__ b,
                                    float* __restrict__ xc, int nc)
{
  const int idx = blockIdx.x * blockDim.x + threadIdx.x;
  if (idx >= nc * 16) return;
  const int node = idx >> 4, o = idx & 15;
  const float* wr = W + o * 44;
  float s = b[o];
  const float* xr = xch + (size_t)node * 12;
#pragma unroll
  for (int k = 0; k < 12; ++k) s = fmaf(xr[k], wr[k], s);
  const float* lr = lstm_last + (size_t)node * 32;
#pragma unroll
  for (int k = 0; k < 32; ++k) s = fmaf(lr[k], wr[12 + k], s);
  xc[idx] = s;
}

__global__ void proj_hru_kernel(const float* __restrict__ xin,
                                const float* __restrict__ W,  // (16,20)
                                const float* __restrict__ b,
                                float* __restrict__ xh, int nh)
{
  const int idx = blockIdx.x * blockDim.x + threadIdx.x;
  if (idx >= nh * 16) return;
  const int node = idx >> 4, o = idx & 15;
  const float* wr = W + o * 20;
  float s = b[o];
  const float* xr = xin + (size_t)node * 20;
#pragma unroll
  for (int k = 0; k < 20; ++k) s = fmaf(xr[k], wr[k], s);
  xh[idx] = s;
}

// ---------------------------------------------------------------------------
// segment-sum scatter: one thread per (edge, feature); f==0 thread bumps cnt
// ---------------------------------------------------------------------------
__global__ void sage_scatter_kernel(const float* __restrict__ x,
                                    const int* __restrict__ ei, long long E,
                                    float* __restrict__ agg,
                                    float* __restrict__ cnt)
{
  const long long idx = (long long)blockIdx.x * blockDim.x + threadIdx.x;
  if (idx >= E * 16) return;
  const long long e = idx >> 4;
  const int f = (int)(idx & 15);
  const int src = ei[e];
  const int dst = ei[E + e];
  atomicAdd(&agg[(size_t)dst * 16 + f], x[(size_t)src * 16 + f]);
  if (f == 0) atomicAdd(&cnt[dst], 1.0f);
}

// ---------------------------------------------------------------------------
// h = mean@Wl^T + bl + x@Wr^T ; out = relu(h@Wo^T + bo)       (one thread/node)
// ---------------------------------------------------------------------------
__global__ void sage_out_kernel(const float* __restrict__ x,
                                const float* __restrict__ agg,
                                const float* __restrict__ cnt,
                                const float* __restrict__ Wl,
                                const float* __restrict__ bl,
                                const float* __restrict__ Wr,
                                const float* __restrict__ Wo,
                                const float* __restrict__ bo,
                                float* __restrict__ out, int nvert)
{
  const int v = blockIdx.x * blockDim.x + threadIdx.x;
  if (v >= nvert) return;
  const float inv = __builtin_amdgcn_rcpf(fmaxf(cnt[v], 1.0f));
  float m[16], xr[16], h[16];
#pragma unroll
  for (int f = 0; f < 16; ++f) {
    m[f]  = agg[(size_t)v * 16 + f] * inv;
    xr[f] = x[(size_t)v * 16 + f];
  }
#pragma unroll
  for (int o = 0; o < 16; ++o) {
    float s = bl[o];
#pragma unroll
    for (int k = 0; k < 16; ++k)
      s = fmaf(m[k], Wl[o * 16 + k], fmaf(xr[k], Wr[o * 16 + k], s));
    h[o] = s;
  }
#pragma unroll
  for (int o = 0; o < 8; ++o) {
    float s = bo[o];
#pragma unroll
    for (int k = 0; k < 16; ++k) s = fmaf(h[k], Wo[o * 16 + k], s);
    out[(size_t)v * 8 + o] = fmaxf(s, 0.0f);
  }
}

// ---------------------------------------------------------------------------
extern "C" void kernel_launch(void* const* d_in, const int* in_sizes, int n_in,
                              void* d_out, int out_size, void* d_ws,
                              size_t ws_size, hipStream_t stream)
{
  const float* x_channel = (const float*)d_in[0];
  const float* x_hru     = (const float*)d_in[1];
  const float* cts       = (const float*)d_in[2];
  const int*   ei_c      = (const int*)d_in[3];
  const int*   ei_h      = (const int*)d_in[4];
  const float* Wih0      = (const float*)d_in[5];
  const float* Whh0      = (const float*)d_in[6];
  const float* bih0      = (const float*)d_in[7];
  const float* bhh0      = (const float*)d_in[8];
  const float* Wih1      = (const float*)d_in[9];
  const float* Whh1      = (const float*)d_in[10];
  const float* bih1      = (const float*)d_in[11];
  const float* bhh1      = (const float*)d_in[12];
  const float* proj_c_W  = (const float*)d_in[13];
  const float* proj_c_b  = (const float*)d_in[14];
  const float* proj_h_W  = (const float*)d_in[15];
  const float* proj_h_b  = (const float*)d_in[16];
  const float* sage_c_Wl = (const float*)d_in[17];
  const float* sage_c_bl = (const float*)d_in[18];
  const float* sage_c_Wr = (const float*)d_in[19];
  const float* sage_h_Wl = (const float*)d_in[20];
  const float* sage_h_bl = (const float*)d_in[21];
  const float* sage_h_Wr = (const float*)d_in[22];
  const float* out_W     = (const float*)d_in[23];
  const float* out_b     = (const float*)d_in[24];

  const int nc = in_sizes[0] / 12;
  const int nh = in_sizes[1] / 20;
  const long long ec = in_sizes[3] / 2;
  const long long eh = in_sizes[4] / 2;
  const int tsteps = in_sizes[2] / (nc * 6);

  float* ws = (float*)d_ws;
  float* lstm_last = ws;  ws += (size_t)nc * 32;
  float* xc        = ws;  ws += (size_t)nc * 16;
  float* xh        = ws;  ws += (size_t)nh * 16;
  float* agg_c     = ws;  ws += (size_t)nc * 16;
  float* cnt_c     = ws;  ws += (size_t)nc;
  float* agg_h     = ws;  ws += (size_t)nh * 16;
  float* cnt_h     = ws;  ws += (size_t)nh;

  // zero the accumulation region (agg_c .. cnt_h is contiguous)
  const size_t zbytes = ((size_t)nc * 17 + (size_t)nh * 17) * sizeof(float);
  hipMemsetAsync(agg_c, 0, zbytes, stream);

  // 1) fused 2-layer LSTM with WMMA (one wave per 16-channel tile)
  const int ntiles = (nc + 15) / 16;
  const int nblk   = (ntiles + 3) / 4;
  lstm2_wmma_kernel<<<nblk, 128, 0, stream>>>(cts, Wih0, Whh0, bih0, bhh0,
                                              Wih1, Whh1, bih1, bhh1,
                                              lstm_last, nc, tsteps);

  const int tpb = 256;
  // 2) projections
  proj_channel_kernel<<<(unsigned)(((size_t)nc * 16 + tpb - 1) / tpb), tpb, 0,
                        stream>>>(x_channel, lstm_last, proj_c_W, proj_c_b, xc, nc);
  proj_hru_kernel<<<(unsigned)(((size_t)nh * 16 + tpb - 1) / tpb), tpb, 0,
                    stream>>>(x_hru, proj_h_W, proj_h_b, xh, nh);

  // 3) SAGE scatter-mean (atomics)
  sage_scatter_kernel<<<(unsigned)((ec * 16 + tpb - 1) / tpb), tpb, 0, stream>>>(
      xc, ei_c, ec, agg_c, cnt_c);
  sage_scatter_kernel<<<(unsigned)((eh * 16 + tpb - 1) / tpb), tpb, 0, stream>>>(
      xh, ei_h, eh, agg_h, cnt_h);

  // 4) combine + output projection + relu
  float* oc = (float*)d_out;
  float* oh = oc + (size_t)nc * 8;
  sage_out_kernel<<<(unsigned)((nc + tpb - 1) / tpb), tpb, 0, stream>>>(
      xc, agg_c, cnt_c, sage_c_Wl, sage_c_bl, sage_c_Wr, out_W, out_b, oc, nc);
  sage_out_kernel<<<(unsigned)((nh + tpb - 1) / tpb), tpb, 0, stream>>>(
      xh, agg_h, cnt_h, sage_h_Wl, sage_h_bl, sage_h_Wr, out_W, out_b, oh, nh);
}
